// Qwen3MoeSparseMoeBlock_40553081208985
// MI455X (gfx1250) — matrix operand order repaired and checked by reading, not compiled, not run on hardware
//
#include <hip/hip_runtime.h>
#include <hip/hip_bf16.h>

typedef __attribute__((ext_vector_type(16))) __bf16 v16bf;
typedef __attribute__((ext_vector_type(8)))  float  v8f;

#define T_TOK 1024
#define H_DIM 2048
#define E_NUM 16
#define I_DIM 768

#define TM 128      // token rows per block tile (covers an average expert load)
#define TN 64       // output columns per block tile
#define TK 32       // K step (one bf16 WMMA depth)

// ---------------------------------------------------------------------------
// Zero output accumulation region + expert counts
// ---------------------------------------------------------------------------
__global__ void zero_kernel(float* __restrict__ out, int n, int* __restrict__ cnt) {
    int i = blockIdx.x * blockDim.x + threadIdx.x;
    int stride = gridDim.x * blockDim.x;
    for (int j = i; j < n; j += stride) out[j] = 0.0f;
    if (blockIdx.x == 0 && threadIdx.x < E_NUM) cnt[threadIdx.x] = 0;
}

// ---------------------------------------------------------------------------
// Router: logits, softmax, top-2, normalized weights, per-expert token lists
// ---------------------------------------------------------------------------
__global__ __launch_bounds__(256) void router_kernel(
    const float* __restrict__ x, const float* __restrict__ gw,
    float* __restrict__ logits_out,
    int* __restrict__ cnt, int* __restrict__ tok_list, float* __restrict__ w_list)
{
    const int t   = blockIdx.x;
    const int tid = threadIdx.x;

    float part[E_NUM];
#pragma unroll
    for (int e = 0; e < E_NUM; ++e) part[e] = 0.0f;

    const float* xr = x + (size_t)t * H_DIM;
    for (int h = tid; h < H_DIM; h += 256) {
        float xv = xr[h];
#pragma unroll
        for (int e = 0; e < E_NUM; ++e) part[e] += xv * gw[e * H_DIM + h];
    }

    // wave32 shuffle reduction
#pragma unroll
    for (int e = 0; e < E_NUM; ++e) {
        for (int off = 16; off > 0; off >>= 1)
            part[e] += __shfl_down(part[e], off, 32);
    }

    __shared__ float red[8][E_NUM];
    __shared__ float lg[E_NUM];
    const int wave = tid >> 5, lane = tid & 31;
    if (lane == 0) {
#pragma unroll
        for (int e = 0; e < E_NUM; ++e) red[wave][e] = part[e];
    }
    __syncthreads();

    if (tid < E_NUM) {
        float s = 0.0f;
#pragma unroll
        for (int w = 0; w < 8; ++w) s += red[w][tid];
        lg[tid] = s;
        logits_out[t * E_NUM + tid] = s;   // raw router logits (2nd output)
    }
    __syncthreads();

    if (tid == 0) {
        float mx = lg[0];
#pragma unroll
        for (int e = 1; e < E_NUM; ++e) mx = fmaxf(mx, lg[e]);
        float p[E_NUM]; float se = 0.0f;
#pragma unroll
        for (int e = 0; e < E_NUM; ++e) { p[e] = __expf(lg[e] - mx); se += p[e]; }
        int i1 = 0;
#pragma unroll
        for (int e = 1; e < E_NUM; ++e) if (p[e] > p[i1]) i1 = e;
        int i2 = (i1 == 0) ? 1 : 0;
#pragma unroll
        for (int e = 0; e < E_NUM; ++e) if (e != i1 && p[e] > p[i2]) i2 = e;
        float p1 = p[i1] / se, p2 = p[i2] / se;
        float inv = 1.0f / (p1 + p2);           // NORM_TOPK_PROB
        float w1v = p1 * inv, w2v = p2 * inv;

        int pos1 = atomicAdd(&cnt[i1], 1);
        tok_list[i1 * T_TOK + pos1] = t;  w_list[i1 * T_TOK + pos1] = w1v;
        int pos2 = atomicAdd(&cnt[i2], 1);
        tok_list[i2 * T_TOK + pos2] = t;  w_list[i2 * T_TOK + pos2] = w2v;
    }
}

// ---------------------------------------------------------------------------
// Prefix offsets over expert counts (compact act-row base)
// ---------------------------------------------------------------------------
__global__ void scan_kernel(const int* __restrict__ cnt, int* __restrict__ off) {
    if (threadIdx.x == 0) {
        int s = 0;
        for (int e = 0; e < E_NUM; ++e) { off[e] = s; s += cnt[e]; }
    }
}

// ---------------------------------------------------------------------------
// Fragment gather helpers (ISA 16-bit A 16x32 / B 32x16 lane layouts)
// ---------------------------------------------------------------------------
__device__ __forceinline__ v16bf load_a_frag(const __bf16 lA[][TK + 2], int mbase, int lane) {
    const int r = lane & 15, kg = lane >> 4;
    v16bf f;
#pragma unroll
    for (int v = 0; v < 8; ++v) {
        int bk = ((v < 4) ? 0 : 16) + kg * 8 + (v & 3) * 2;
        f[2 * v]     = lA[mbase + r][bk];
        f[2 * v + 1] = lA[mbase + r][bk + 1];
    }
    return f;
}

__device__ __forceinline__ v16bf load_b_frag(const __bf16 lB[][TK + 2], int nbase, int lane) {
    const int r = lane & 15, kg = lane >> 4;
    v16bf f;
#pragma unroll
    for (int v = 0; v < 8; ++v) {
        int kk = ((v < 4) ? 0 : 16) + kg * 8 + (v & 3) * 2;
        f[2 * v]     = lB[nbase + r][kk];
        f[2 * v + 1] = lB[nbase + r][kk + 1];
    }
    return f;
}

// ---------------------------------------------------------------------------
// GEMM1: h = x @ w1_e.T for routed rows; fused SiLU(gate)*up*route_w -> act bf16
// grid.x = I/TN (12), grid.y = E * (T/TM) (128), block = 256 (8 waves)
// Wave = (wgrp in {0,1}) x (wn in {0..3}); each wave: 4 M-subtiles x 1 N-subtile
// ---------------------------------------------------------------------------
__global__ __launch_bounds__(256) void moe_gemm1(
    const float* __restrict__ x, const float* __restrict__ w1,
    const int* __restrict__ cnt, const int* __restrict__ off,
    const int* __restrict__ tok_list, const float* __restrict__ w_list,
    __bf16* __restrict__ act)
{
    const int e     = blockIdx.y >> 3;       // 8 token tiles per expert
    const int tile  = blockIdx.y & 7;
    const int n0    = blockIdx.x * TN;
    const int count = cnt[e];
    const int row0  = tile * TM;
    if (row0 >= count) return;               // block-uniform: EXEC stays all-ones
    const int base  = off[e];

    __shared__ __bf16 lA [TM][TK + 2];
    __shared__ __bf16 lBg[TN][TK + 2];
    __shared__ __bf16 lBu[TN][TK + 2];

    const int tid  = threadIdx.x;
    const int arow = tid >> 3;               // 0..31
    const int acol = (tid & 7) * 4;          // 0,4,..,28

    // A: 4 gathered token rows per thread (arow + q*32)
    const float* aptr[4];
#pragma unroll
    for (int q = 0; q < 4; ++q) {
        int rg  = row0 + arow + q * 32;
        int tok = tok_list[e * T_TOK + min(rg, count - 1)];
        aptr[q] = x + (size_t)tok * H_DIM + acol;
    }

    const float* w1e = w1 + (size_t)e * (2 * I_DIM) * H_DIM;
    const float* bg0 = w1e + (size_t)(n0 + arow)              * H_DIM + acol;
    const float* bg1 = w1e + (size_t)(n0 + arow + 32)         * H_DIM + acol;
    const float* bu0 = w1e + (size_t)(I_DIM + n0 + arow)      * H_DIM + acol;
    const float* bu1 = w1e + (size_t)(I_DIM + n0 + arow + 32) * H_DIM + acol;

    const int wave = tid >> 5;
    const int lane = tid & 31;
    const int wn   = wave & 3;               // N-subtile
    const int wgrp = wave >> 2;              // M half (rows wgrp*64 .. +63)

    v8f accg[4], accu[4];
#pragma unroll
    for (int j = 0; j < 4; ++j) { accg[j] = (v8f){}; accu[j] = (v8f){}; }

    for (int k0 = 0; k0 < H_DIM; k0 += TK) {
        float4 av[4];
#pragma unroll
        for (int q = 0; q < 4; ++q) av[q] = *(const float4*)(aptr[q] + k0);
        float4 g0 = *(const float4*)(bg0 + k0);
        float4 g1 = *(const float4*)(bg1 + k0);
        float4 u0 = *(const float4*)(bu0 + k0);
        float4 u1 = *(const float4*)(bu1 + k0);
        // stream hints for the next weight chunk (speculative; OOB drops)
        __builtin_prefetch(bg0 + k0 + TK, 0, 3);
        __builtin_prefetch(bg1 + k0 + TK, 0, 3);
        __builtin_prefetch(bu0 + k0 + TK, 0, 3);
        __builtin_prefetch(bu1 + k0 + TK, 0, 3);

        __syncthreads();
#pragma unroll
        for (int q = 0; q < 4; ++q) {
            lA[arow + q * 32][acol + 0] = (__bf16)av[q].x;
            lA[arow + q * 32][acol + 1] = (__bf16)av[q].y;
            lA[arow + q * 32][acol + 2] = (__bf16)av[q].z;
            lA[arow + q * 32][acol + 3] = (__bf16)av[q].w;
        }
        lBg[arow][acol + 0] = (__bf16)g0.x; lBg[arow][acol + 1] = (__bf16)g0.y;
        lBg[arow][acol + 2] = (__bf16)g0.z; lBg[arow][acol + 3] = (__bf16)g0.w;
        lBg[arow + 32][acol + 0] = (__bf16)g1.x; lBg[arow + 32][acol + 1] = (__bf16)g1.y;
        lBg[arow + 32][acol + 2] = (__bf16)g1.z; lBg[arow + 32][acol + 3] = (__bf16)g1.w;
        lBu[arow][acol + 0] = (__bf16)u0.x; lBu[arow][acol + 1] = (__bf16)u0.y;
        lBu[arow][acol + 2] = (__bf16)u0.z; lBu[arow][acol + 3] = (__bf16)u0.w;
        lBu[arow + 32][acol + 0] = (__bf16)u1.x; lBu[arow + 32][acol + 1] = (__bf16)u1.y;
        lBu[arow + 32][acol + 2] = (__bf16)u1.z; lBu[arow + 32][acol + 3] = (__bf16)u1.w;
        __syncthreads();

        // B fragments loaded once, reused across 4 M-subtiles
        v16bf bgf = load_b_frag(lBg, wn * 16, lane);
        v16bf buf = load_b_frag(lBu, wn * 16, lane);
#pragma unroll
        for (int j = 0; j < 4; ++j) {
            v16bf af = load_a_frag(lA, wgrp * 64 + j * 16, lane);
            accg[j] = __builtin_amdgcn_wmma_f32_16x16x32_bf16(false, af, false, bgf,
                                                              (short)0, accg[j], false, false);
            accu[j] = __builtin_amdgcn_wmma_f32_16x16x32_bf16(false, af, false, buf,
                                                              (short)0, accu[j], false, false);
        }
    }

    // epilogue: act = silu(gate) * up * routing_weight  (weight folded here)
    const int cn  = lane & 15;
    const int chi = lane >> 4;
#pragma unroll
    for (int j = 0; j < 4; ++j) {
#pragma unroll
        for (int rr = 0; rr < 8; ++rr) {
            int m    = wgrp * 64 + j * 16 + rr + 8 * chi;
            int grow = row0 + m;
            if (grow < count) {
                float g  = accg[j][rr], u = accu[j][rr];
                float sv = g / (1.0f + __expf(-g));
                float rw = w_list[e * T_TOK + grow];
                act[(size_t)(base + grow) * I_DIM + n0 + wn * 16 + cn] = (__bf16)(sv * u * rw);
            }
        }
    }
}

// ---------------------------------------------------------------------------
// GEMM2: out[tok] += act_row @ w2_e.T   (atomic scatter over top-2 collisions)
// grid.x = H/TN (32), grid.y = E * (T/TM) (128), block = 256
// ---------------------------------------------------------------------------
__global__ __launch_bounds__(256) void moe_gemm2(
    const __bf16* __restrict__ act, const float* __restrict__ w2,
    const int* __restrict__ cnt, const int* __restrict__ off,
    const int* __restrict__ tok_list,
    float* __restrict__ out)
{
    const int e     = blockIdx.y >> 3;
    const int tile  = blockIdx.y & 7;
    const int n0    = blockIdx.x * TN;
    const int count = cnt[e];
    const int row0  = tile * TM;
    if (row0 >= count) return;
    const int base  = off[e];

    __shared__ __bf16 lA[TM][TK + 2];
    __shared__ __bf16 lB[TN][TK + 2];

    const int tid  = threadIdx.x;
    const int arow = tid >> 3;
    const int acol = (tid & 7) * 4;

    const __bf16* aptr[4];
#pragma unroll
    for (int q = 0; q < 4; ++q) {
        int rg = min(row0 + arow + q * 32, count - 1);
        aptr[q] = act + (size_t)(base + rg) * I_DIM + acol;
    }

    const float* w2e = w2 + (size_t)e * H_DIM * I_DIM;
    const float* b0  = w2e + (size_t)(n0 + arow)      * I_DIM + acol;
    const float* b1  = w2e + (size_t)(n0 + arow + 32) * I_DIM + acol;

    const int wave = tid >> 5;
    const int lane = tid & 31;
    const int wn   = wave & 3;
    const int wgrp = wave >> 2;

    v8f acc[4];
#pragma unroll
    for (int j = 0; j < 4; ++j) acc[j] = (v8f){};

    for (int k0 = 0; k0 < I_DIM; k0 += TK) {
        union { uint2 v; __bf16 h[4]; } au[4];
#pragma unroll
        for (int q = 0; q < 4; ++q) au[q].v = *(const uint2*)(aptr[q] + k0);
        float4 v0 = *(const float4*)(b0 + k0);
        float4 v1 = *(const float4*)(b1 + k0);
        __builtin_prefetch(b0 + k0 + TK, 0, 3);
        __builtin_prefetch(b1 + k0 + TK, 0, 3);

        __syncthreads();
#pragma unroll
        for (int q = 0; q < 4; ++q) {
            lA[arow + q * 32][acol + 0] = au[q].h[0];
            lA[arow + q * 32][acol + 1] = au[q].h[1];
            lA[arow + q * 32][acol + 2] = au[q].h[2];
            lA[arow + q * 32][acol + 3] = au[q].h[3];
        }
        lB[arow][acol + 0] = (__bf16)v0.x; lB[arow][acol + 1] = (__bf16)v0.y;
        lB[arow][acol + 2] = (__bf16)v0.z; lB[arow][acol + 3] = (__bf16)v0.w;
        lB[arow + 32][acol + 0] = (__bf16)v1.x; lB[arow + 32][acol + 1] = (__bf16)v1.y;
        lB[arow + 32][acol + 2] = (__bf16)v1.z; lB[arow + 32][acol + 3] = (__bf16)v1.w;
        __syncthreads();

        v16bf bf = load_b_frag(lB, wn * 16, lane);
#pragma unroll
        for (int j = 0; j < 4; ++j) {
            v16bf af = load_a_frag(lA, wgrp * 64 + j * 16, lane);
            acc[j] = __builtin_amdgcn_wmma_f32_16x16x32_bf16(false, af, false, bf,
                                                             (short)0, acc[j], false, false);
        }
    }

    const int cn  = lane & 15;
    const int chi = lane >> 4;
#pragma unroll
    for (int j = 0; j < 4; ++j) {
#pragma unroll
        for (int rr = 0; rr < 8; ++rr) {
            int m    = wgrp * 64 + j * 16 + rr + 8 * chi;
            int grow = row0 + m;
            if (grow < count) {
                int tok = tok_list[e * T_TOK + grow];
                atomicAdd(&out[(size_t)tok * H_DIM + n0 + wn * 16 + cn], acc[j][rr]);
            }
        }
    }
}

// ---------------------------------------------------------------------------
extern "C" void kernel_launch(void* const* d_in, const int* in_sizes, int n_in,
                              void* d_out, int out_size, void* d_ws, size_t ws_size,
                              hipStream_t stream) {
    const float* x  = (const float*)d_in[0];   // [1,1024,2048]
    const float* gw = (const float*)d_in[1];   // [16,2048]
    const float* w1 = (const float*)d_in[2];   // [16,1536,2048]
    const float* w2 = (const float*)d_in[3];   // [16,2048,768]

    float* out    = (float*)d_out;               // [1024*2048]
    float* logits = out + (size_t)T_TOK * H_DIM; // [1024*16]

    // workspace layout
    int*    cnt      = (int*)d_ws;                         // 16
    int*    off      = cnt + E_NUM;                        // 16
    int*    tok_list = off + E_NUM;                        // 16*1024
    float*  w_list   = (float*)(tok_list + E_NUM * T_TOK); // 16*1024
    __bf16* act      = (__bf16*)(w_list + E_NUM * T_TOK);  // 2048*768 bf16

    zero_kernel<<<512, 256, 0, stream>>>(out, T_TOK * H_DIM, cnt);
    router_kernel<<<T_TOK, 256, 0, stream>>>(x, gw, logits, cnt, tok_list, w_list);
    scan_kernel<<<1, 32, 0, stream>>>(cnt, off);

    dim3 g1(I_DIM / TN, E_NUM * (T_TOK / TM));   // (12, 128)
    moe_gemm1<<<g1, 256, 0, stream>>>(x, w1, cnt, off, tok_list, w_list, act);

    dim3 g2(H_DIM / TN, E_NUM * (T_TOK / TM));   // (32, 128)
    moe_gemm2<<<g2, 256, 0, stream>>>(act, w2, cnt, off, tok_list, out);
}
